// Head_21139829031061
// MI455X (gfx1250) — matrix operand order repaired and checked by reading, not compiled, not run on hardware
//
#include <hip/hip_runtime.h>

typedef __attribute__((ext_vector_type(16))) __bf16 v16bf;
typedef __attribute__((ext_vector_type(8)))  __bf16 v8bf;
typedef __attribute__((ext_vector_type(8)))  float  v8f;

#define B_  8
#define T_  2048
#define C_  1024
#define H_  128
#define NKB (T_ / 64)          // 32 key blocks of 64 per batch

// Fragment-major workspace layouts (bf16): addr = tile*512 + lane*16 + j, so
// each lane's v16bf WMMA operand is one contiguous 32-byte chunk.
//   Q: A-frags, tile = (m/16)*4 + h/32
//   K: B-frags for QK^T, per 64-key block 16 tiles (ks4 x tt4)
//   V: B-frags for PV,   per 64-key block 16 tiles (ks2 x t8)

__device__ __forceinline__ unsigned lds_off(const void* p) {
  return (unsigned)(unsigned long long)(uintptr_t)p;   // low 32 bits = LDS byte offset
}
// CDNA5 async global->LDS copy (ASYNCcnt-tracked), 16 bytes per lane.
__device__ __forceinline__ void async_cp16(const __bf16* g, __bf16* l) {
  asm volatile("global_load_async_to_lds_b128 %0, %1, off"
               :: "v"(lds_off(l)), "v"((unsigned long long)(uintptr_t)g)
               : "memory");
}

// ---------------------------------------------------------------------------
// Projection GEMM: y = x @ W (M=16384, K=1024, N=128), bf16 WMMA, f32 acc.
// blockIdx.y: 0=Q (scaled, A-frag layout), 1=K (B-frag), 2=V (B-frag).
// Epilogue repacks C-tiles through per-wave LDS so all global stores are b128.
// ---------------------------------------------------------------------------
__global__ __launch_bounds__(256) void proj_kernel(
    const float* __restrict__ x,  const float* __restrict__ Wk,
    const float* __restrict__ Wq, const float* __restrict__ Wv,
    __bf16* __restrict__ Qs, __bf16* __restrict__ Ks, __bf16* __restrict__ Vs)
{
  __shared__ __align__(16) __bf16 lx[8 * 512];    // 8 A-tiles (128 rows x 32 k)
  __shared__ __align__(16) __bf16 lw[8 * 512];    // 8 B-tiles (32 k x 128 cols)
  __shared__ __align__(16) __bf16 rep[8][2048];   // per-wave repack (4 KB each)

  const int tid  = threadIdx.x;
  const int lane = tid & 31;
  const int w    = tid >> 5;
  const int mode = blockIdx.y;
  const float* W = (mode == 0) ? Wq : (mode == 1) ? Wk : Wv;
  const int row0 = blockIdx.x * 128;

  const int xrow  = tid >> 1;          // 0..127
  const int xhalf = tid & 1;           // which 16-float half of the 32-k tile
  const int wcol  = tid & 127;         // W column
  const int wg0   = tid >> 7;          // W k-groups {wg0, wg0+2}

  v8f zero = {};
  v8f acc[8];
  #pragma unroll
  for (int t = 0; t < 8; ++t) acc[t] = zero;

  for (int kk = 0; kk < C_; kk += 32) {
    // ---- stage x tile (pre-swizzled to A-fragment layout) ----
    {
      const float4* x4 = (const float4*)(x + (size_t)(row0 + xrow) * C_ + kk + xhalf * 16);
      float4 f0 = x4[0], f1 = x4[1], f2 = x4[2], f3 = x4[3];
      v8bf p0, p1;
      p0[0] = (__bf16)f0.x; p0[1] = (__bf16)f0.y; p0[2] = (__bf16)f0.z; p0[3] = (__bf16)f0.w;
      p0[4] = (__bf16)f1.x; p0[5] = (__bf16)f1.y; p0[6] = (__bf16)f1.z; p0[7] = (__bf16)f1.w;
      p1[0] = (__bf16)f2.x; p1[1] = (__bf16)f2.y; p1[2] = (__bf16)f2.z; p1[3] = (__bf16)f2.w;
      p1[4] = (__bf16)f3.x; p1[5] = (__bf16)f3.y; p1[6] = (__bf16)f3.z; p1[7] = (__bf16)f3.w;
      int rt = xrow >> 4, r16 = xrow & 15;
      int base = rt * 512 + r16 * 16 + xhalf * 8;
      *(v8bf*)&lx[base]       = p0;    // k-halves land in lanes r16 / r16+16
      *(v8bf*)&lx[base + 256] = p1;
    }
    // ---- stage W tile (pre-swizzled to B-fragment layout) ----
    #pragma unroll
    for (int gg = 0; gg < 2; ++gg) {
      int g = wg0 + 2 * gg;            // k-group of 8: k = g*8..g*8+7
      v8bf pw;
      #pragma unroll
      for (int u = 0; u < 8; ++u)
        pw[u] = (__bf16)W[(size_t)(kk + g * 8 + u) * H_ + wcol];
      int t = wcol >> 4, nl = wcol & 15, hig = g & 1, j0 = (g >> 1) * 8;
      *(v8bf*)&lw[t * 512 + (hig * 16 + nl) * 16 + j0] = pw;
    }
    __syncthreads();

    v16bf a = *(const v16bf*)&lx[w * 512 + lane * 16];
    #pragma unroll
    for (int t = 0; t < 8; ++t) {
      v16bf bb = *(const v16bf*)&lw[t * 512 + lane * 16];
      acc[t] = __builtin_amdgcn_wmma_f32_16x16x32_bf16(
          false, a, false, bb, (short)0, acc[t], false, false);
    }
    __syncthreads();
  }

  // ---- epilogue: scatter into per-wave LDS (destination order), then
  //      wave-cooperative linear copy-out with wide stores ----
  const int hi = (lane >> 4) & 1;
  const int nl = lane & 15;
  const int m0 = row0 + 16 * w;                   // wave's first global row
  __bf16* R = &rep[w][0];

  if (mode == 0) {
    const float qscale = 0.08838834764831845f;    // 1/sqrt(128)
    #pragma unroll
    for (int t = 0; t < 8; ++t) {
      #pragma unroll
      for (int r = 0; r < 8; ++r) {
        int row16 = r + 8 * hi;
        int col   = 16 * t + nl;
        int laneA = (((col >> 4) & 1) << 4) | row16;
        int jA    = col & 15;
        R[((col >> 5) << 9) + laneA * 16 + jA] = (__bf16)(acc[t][r] * qscale);
      }
    }
    asm volatile("s_wait_dscnt 0" ::: "memory");
    __bf16* g = Qs + ((size_t)(m0 >> 4)) * 4 * 512;   // 4 contiguous tiles
    #pragma unroll
    for (int c = 0; c < 4; ++c)
      *(v16bf*)&g[c * 512 + lane * 16] = *(const v16bf*)&R[c * 512 + lane * 16];
  } else if (mode == 1) {
    const int bb = m0 >> 11, key0 = m0 & (T_ - 1);
    const int kb = key0 >> 6, tt = (key0 >> 4) & 3;
    #pragma unroll
    for (int t = 0; t < 8; ++t) {
      #pragma unroll
      for (int r = 0; r < 8; ++r) {
        int row16 = r + 8 * hi;                   // key & 15
        R[(t >> 1) * 512 + ((t & 1) * 16 + row16) * 16 + nl] = (__bf16)acc[t][r];
      }
    }
    asm volatile("s_wait_dscnt 0" ::: "memory");
    size_t base = ((size_t)(bb * NKB + kb) * 16 + tt) * 512;   // tile ks*4+tt
    #pragma unroll
    for (int c = 0; c < 4; ++c)
      *(v16bf*)&Ks[base + c * 4 * 512 + lane * 16] = *(const v16bf*)&R[c * 512 + lane * 16];
  } else {
    const int bb = m0 >> 11, key0 = m0 & (T_ - 1);
    const int kb = key0 >> 6;
    const int ksv = (key0 >> 5) & 1, hv = (key0 >> 4) & 1;    // wave-uniform
    #pragma unroll
    for (int t = 0; t < 8; ++t) {
      #pragma unroll
      for (int r = 0; r < 8; ++r) {
        int row16 = r + 8 * hi;                   // key & 15 -> j
        R[t * 256 + nl * 16 + row16] = (__bf16)acc[t][r];
      }
    }
    asm volatile("s_wait_dscnt 0" ::: "memory");
    size_t base = ((size_t)(bb * NKB + kb) * 16 + ksv * 8) * 512 + hv * 256;
    #pragma unroll
    for (int t = 0; t < 8; ++t)
      *(v8bf*)&Vs[base + t * 512 + lane * 8] = *(const v8bf*)&R[t * 256 + lane * 8];
  }
}

// ---------------------------------------------------------------------------
// Fused causal flash attention. Block = 128 query rows (8 waves x 16 rows).
// Key blocks of 64 staged to LDS with double-buffered async copies.
// ---------------------------------------------------------------------------
__global__ __launch_bounds__(256) void attn_kernel(
    const __bf16* __restrict__ Qs, const __bf16* __restrict__ Ks,
    const __bf16* __restrict__ Vs, float* __restrict__ out)
{
  __shared__ __align__(16) __bf16 lk[2][16 * 512];   // K B-frags, double buffered
  __shared__ __align__(16) __bf16 lv[2][16 * 512];   // V B-frags, double buffered
  __shared__ __align__(16) __bf16 lp[8][2 * 512];    // per-wave P A-frags

  const int tid  = threadIdx.x;
  const int lane = tid & 31;
  const int w    = tid >> 5;
  const int hi   = (lane >> 4) & 1;
  const int nl   = lane & 15;
  const int b    = blockIdx.y;
  const int q0   = blockIdx.x * 128 + w * 16;

  // hoisted Q A-fragments (contiguous v16bf loads from fragment-major ws)
  const size_t qoff = ((size_t)((b * T_ + q0) >> 4)) * 4 * 512;
  v16bf qa[4];
  #pragma unroll
  for (int ks = 0; ks < 4; ++ks)
    qa[ks] = *(const v16bf*)&Qs[qoff + ks * 512 + lane * 16];

  v8f zero = {};
  v8f o[8];
  #pragma unroll
  for (int t = 0; t < 8; ++t) o[t] = zero;
  float mrow[8], lrow[8];
  #pragma unroll
  for (int r = 0; r < 8; ++r) { mrow[r] = -1e30f; lrow[r] = 0.f; }

  const int nkb_blk = 2 * blockIdx.x + 2;   // block-uniform causal bound
  const int nkb_w   = (q0 >> 6) + 1;        // this wave's causal bound
  __bf16* P = &lp[w][0];

  auto issue = [&](int kb, int d) {         // 8 async b128 ops per lane
    const __bf16* gk = Ks + ((size_t)(b * NKB + kb)) * 8192;
    const __bf16* gv = Vs + ((size_t)(b * NKB + kb)) * 8192;
    #pragma unroll
    for (int u = 0; u < 4; ++u) {
      int c = (tid + 256 * u) * 8;
      async_cp16(gk + c, &lk[d][c]);
      async_cp16(gv + c, &lv[d][c]);
    }
  };

  issue(0, 0);
  for (int kb = 0; kb < nkb_blk; ++kb) {
    const int d = kb & 1;
    const bool more = (kb + 1 < nkb_blk);
    if (more) {
      issue(kb + 1, d ^ 1);                           // prefetch next block
      asm volatile("s_wait_asynccnt 8" ::: "memory"); // current block ready
    } else {
      asm volatile("s_wait_asynccnt 0" ::: "memory");
    }
    __syncthreads();

    if (kb < nkb_w) {
      const int s0 = kb * 64;

      // ---- S = Q K^T : 4 C-tiles, 4 k-steps, all operands contiguous ----
      v8f st[4];
      #pragma unroll
      for (int t = 0; t < 4; ++t) st[t] = zero;
      #pragma unroll
      for (int ks = 0; ks < 4; ++ks) {
        #pragma unroll
        for (int t = 0; t < 4; ++t) {
          v16bf bk = *(const v16bf*)&lk[d][(ks * 4 + t) * 512 + lane * 16];
          st[t] = __builtin_amdgcn_wmma_f32_16x16x32_bf16(
              false, qa[ks], false, bk, (short)0, st[t], false, false);
        }
      }

      // ---- causal mask (C layout: VGPR r -> row, lane -> col) ----
      #pragma unroll
      for (int t = 0; t < 4; ++t) {
        #pragma unroll
        for (int r = 0; r < 8; ++r) {
          int qrow = q0 + r + 8 * hi;
          int kcol = s0 + t * 16 + nl;
          if (kcol > qrow) st[t][r] = -1e30f;
        }
      }

      // ---- online softmax; P written directly in A-fragment layout ----
      #pragma unroll
      for (int r = 0; r < 8; ++r) {
        float mx = fmaxf(fmaxf(st[0][r], st[1][r]), fmaxf(st[2][r], st[3][r]));
        #pragma unroll
        for (int dd = 1; dd < 16; dd <<= 1) mx = fmaxf(mx, __shfl_xor(mx, dd, 32));
        float mnew = fmaxf(mrow[r], mx);
        float corr = __expf(mrow[r] - mnew);
        float psum = 0.f;
        int row16 = r + 8 * hi;
        #pragma unroll
        for (int t = 0; t < 4; ++t) {
          float p = __expf(st[t][r] - mnew);
          psum += p;
          int col   = t * 16 + nl;
          int laneA = (((col >> 4) & 1) << 4) | row16;
          int jA    = col & 15;
          P[((col >> 5) << 9) + laneA * 16 + jA] = (__bf16)p;
        }
        #pragma unroll
        for (int dd = 1; dd < 16; dd <<= 1) psum += __shfl_xor(psum, dd, 32);
        lrow[r] = lrow[r] * corr + psum;
        mrow[r] = mnew;
        #pragma unroll
        for (int t = 0; t < 8; ++t) o[t][r] *= corr;
      }

      // cross-lane LDS RAW inside the wave before re-reading P
      asm volatile("s_wait_dscnt 0" ::: "memory");

      // ---- O += P V : 2 k-steps x 8 n-tiles, contiguous operands ----
      #pragma unroll
      for (int ks = 0; ks < 2; ++ks) {
        v16bf pa = *(const v16bf*)&P[ks * 512 + lane * 16];
        #pragma unroll
        for (int t = 0; t < 8; ++t) {
          v16bf bv = *(const v16bf*)&lv[d][(ks * 8 + t) * 512 + lane * 16];
          o[t] = __builtin_amdgcn_wmma_f32_16x16x32_bf16(
              false, pa, false, bv, (short)0, o[t], false, false);
        }
      }
    }
    __syncthreads();   // all waves done with buffer d before it is refilled
  }

  // ---- normalize and write f32 output [B,T,H] ----
  #pragma unroll
  for (int t = 0; t < 8; ++t) {
    #pragma unroll
    for (int r = 0; r < 8; ++r) {
      int qrow  = q0 + r + 8 * hi;
      float inv = 1.f / lrow[r];
      out[((size_t)b * T_ + qrow) * H_ + t * 16 + nl] = o[t][r] * inv;
    }
  }
}

extern "C" void kernel_launch(void* const* d_in, const int* in_sizes, int n_in,
                              void* d_out, int out_size, void* d_ws, size_t ws_size,
                              hipStream_t stream) {
  const float* x  = (const float*)d_in[0];
  const float* Wk = (const float*)d_in[1];
  const float* Wq = (const float*)d_in[2];
  const float* Wv = (const float*)d_in[3];
  float* out = (float*)d_out;

  const size_t per = (size_t)B_ * T_ * H_;   // elements per bf16 tensor
  __bf16* Qs = (__bf16*)d_ws;                // 4 MB, A-fragment major
  __bf16* Ks = Qs + per;                     // 4 MB, B-fragment major (QK^T)
  __bf16* Vs = Ks + per;                     // 4 MB, B-fragment major (PV)

  dim3 g1(B_ * T_ / 128, 3);
  proj_kernel<<<g1, 256, 0, stream>>>(x, Wk, Wq, Wv, Qs, Ks, Vs);

  dim3 g2(T_ / 128, B_);
  attn_kernel<<<g2, 256, 0, stream>>>(Qs, Ks, Vs, out);
}
